// Encoder_z0_NFRNN_64587718197719
// MI455X (gfx1250) — matrix operand order repaired and checked by reading, not compiled, not run on hardware
//
#include <hip/hip_runtime.h>

#define DEVI __device__ __forceinline__

typedef __bf16 bf16;
typedef bf16  bf16x8  __attribute__((ext_vector_type(8)));
typedef bf16  bf16x16 __attribute__((ext_vector_type(16)));
typedef float v8f     __attribute__((ext_vector_type(8)));
typedef unsigned int u32x4 __attribute__((ext_vector_type(4)));
typedef int i32x4 __attribute__((ext_vector_type(4)));
typedef int i32x8 __attribute__((ext_vector_type(8)));

// problem sizes
#define B_    4096
#define T_    64
#define RECN  128
#define HIDN  256
#define BM    64      // batch rows per workgroup
#define NW    8       // waves per block (256 threads, wave32)

// bf16 weight workspace layout (element offsets)
#define OFF_F0W0  0         // 256x128
#define OFF_F0W1  32768     // 256x256
#define OFF_F0W2  98304     // 256x256
#define OFF_F1W0  163840
#define OFF_F1W1  196608
#define OFF_F1W2  262144
#define OFF_WIH   327680    // 384x64
#define OFF_WHH   352256    // 384x128
#define OFF_ZW0   401408    // 112x128 (rows 100..111 zero)
#define OFF_ZW1   415744    // 128x128 (cols 100..127 zero)

// Tensor Data Mover availability (arity differs across toolchains)
#ifdef __has_builtin
# if __has_builtin(__builtin_amdgcn_tensor_load_to_lds) && \
     __has_builtin(__builtin_amdgcn_s_wait_tensorcnt)
#  define HAVE_TDM 1
# endif
#endif
#ifndef HAVE_TDM
# define HAVE_TDM 0
#endif

DEVI bf16 tobf(float f) {
  unsigned u = __float_as_uint(f);
  unsigned r = (u + 0x7FFFu + ((u >> 16) & 1u)) >> 16;
  unsigned short s = (unsigned short)r;
  return __builtin_bit_cast(bf16, s);
}

// ---- branch-free transcendentals (native v_exp/v_log/v_rcp/v_tanh) ----
DEVI float fexp2(float x) {
#if __has_builtin(__builtin_amdgcn_exp2f)
  return __builtin_amdgcn_exp2f(x);
#else
  return exp2f(x);
#endif
}
DEVI float flog2(float x) {
#if __has_builtin(__builtin_amdgcn_logf)
  return __builtin_amdgcn_logf(x);
#else
  return log2f(x);
#endif
}
DEVI float frcp(float x) {
#if __has_builtin(__builtin_amdgcn_rcpf)
  return __builtin_amdgcn_rcpf(x);
#else
  return 1.f / x;
#endif
}
DEVI float fast_tanh(float x) {
#if __has_builtin(__builtin_amdgcn_tanhf)
  return __builtin_amdgcn_tanhf(x);          // v_tanh_f32 (CDNA5 trans op)
#else
  float ax = __builtin_fabsf(x);
  float e = fexp2(ax * -2.88539008177793f);  // exp(-2|x|)
  float t = (1.f - e) * frcp(1.f + e);
  return __uint_as_float((__float_as_uint(x) & 0x80000000u) | __float_as_uint(t));
#endif
}
DEVI float fast_sigmoid(float x) { return frcp(1.f + fexp2(x * -1.44269504088896f)); }
DEVI float fast_exp(float x)     { return fexp2(x * 1.44269504088896f); }
DEVI float fast_softplus(float x) {
  float ax = __builtin_fabsf(x);
  float l = flog2(1.f + fexp2(-1.44269504088896f * ax)) * 0.69314718055995f;
  return fmaxf(x, 0.f) + l;
}

// A operand: 16x32 bf16 tile row for this lane from LDS.
// low lanes hold K = [kb..kb+15]; high lanes +16.
DEVI bf16x16 ldsA(const bf16* p) {
  bf16x8 a = *(const bf16x8*)p;
  bf16x8 b = *(const bf16x8*)(p + 8);
  return __builtin_shufflevector(a, b, 0,1,2,3,4,5,6,7,8,9,10,11,12,13,14,15);
}
// B operand: 32x16 bf16; lane n holds 16 contiguous K values of W row n.
DEVI bf16x16 gldB(const bf16* p) {
  bf16x8 a = *(const bf16x8*)p;
  bf16x8 b = *(const bf16x8*)(p + 8);
  return __builtin_shufflevector(a, b, 0,1,2,3,4,5,6,7,8,9,10,11,12,13,14,15);
}
DEVI v8f wmma_bf(bf16x16 a, bf16x16 b, v8f c) {
  return __builtin_amdgcn_wmma_f32_16x16x32_bf16(false, a, false, b, (short)0, c, false, false);
}

// generic tiled GEMM: O = tanh(A[BM,K] * W[N,K]^T + bias), bf16 out to LDS
DEVI void gemm_bias_tanh(const bf16* A, int As, const bf16* W, int K,
                         const float* bias, bf16* O, int Os,
                         int NTt, int Ttot, int wave, int lane) {
  int nl = lane & 15, hi = lane >> 4;
  for (int t = wave; t < Ttot; t += NW) {
    int mt = t / NTt, nt = t - mt * NTt;
    int col = nt * 16 + nl;
    float b = bias ? bias[col] : 0.f;
    v8f acc;
#pragma unroll
    for (int v = 0; v < 8; ++v) acc[v] = b;
#pragma unroll
    for (int kb = 0; kb < K; kb += 32) {
      bf16x16 av = ldsA(A + (mt * 16 + nl) * As + kb + hi * 16);
      bf16x16 bv = gldB(W + col * K + kb + hi * 16);
      acc = wmma_bf(av, bv, acc);
    }
#pragma unroll
    for (int v = 0; v < 8; ++v) {
      int r = mt * 16 + v + hi * 8;
      O[r * Os + col] = tobf(fast_tanh(acc[v]));
    }
  }
}

__global__ void cvt_pad(bf16* __restrict__ dst, const float* __restrict__ src,
                        int src_rows, int src_cols, int src_stride,
                        int dst_rows, int dst_cols) {
  int n = dst_rows * dst_cols;
  for (int e = blockIdx.x * blockDim.x + threadIdx.x; e < n;
       e += gridDim.x * blockDim.x) {
    int r = e / dst_cols, c = e - r * dst_cols;
    float v = (r < src_rows && c < src_cols) ? src[r * src_stride + c] : 0.f;
    dst[e] = tobf(v);
  }
}

__global__ __launch_bounds__(256)
void nfrnn_main(const float* __restrict__ data, const float* __restrict__ tsteps,
                const float* __restrict__ f0w0, const float* __restrict__ f0b0,
                const float* __restrict__ f0b1, const float* __restrict__ f0b2,
                const float* __restrict__ f0tw,
                const float* __restrict__ f1w0, const float* __restrict__ f1b0,
                const float* __restrict__ f1b1, const float* __restrict__ f1b2,
                const float* __restrict__ f1tw,
                const float* __restrict__ gbih, const float* __restrict__ gbhh,
                const float* __restrict__ zb0, const float* __restrict__ zb1,
                const bf16* __restrict__ wsb, float* __restrict__ out) {
  __shared__ float hs[BM * RECN];                       // 32 KB h state (f32)
  __shared__ __align__(16) bf16 abuf[BM * HIDN];        // 32 KB
  __shared__ __align__(16) bf16 obuf[BM * HIDN];        // 32 KB
  __shared__ __align__(16) bf16 xib[BM * 64];           // 8 KB bf16 x_i
  __shared__ __align__(16) float xif[BM * 64];          // 16 KB f32 x_i (TDM dest)
  __shared__ float bias0v[HIDN];
  __shared__ float tvec[HIDN];
  __shared__ float rowm[BM];

  const int tid = threadIdx.x;
  const int lane = tid & 31, wave = tid >> 5;
  const int nl = lane & 15, hi = lane >> 4;
  const int row0 = blockIdx.x * BM;

  for (int e = tid; e < BM * RECN; e += 256) hs[e] = 0.f;
  __syncthreads();

  for (int i = 0; i < T_; ++i) {
    float dt = (i == 0) ? -0.01f : (tsteps[63 - i] - tsteps[64 - i]);

    // ---- stage x_i tile (64 rows x 64 f32, row stride T*64) into LDS ----
#if HAVE_TDM
    if (wave == 0) {
      unsigned long long ga =
          (unsigned long long)(size_t)(data + ((size_t)row0 * T_ + (63 - i)) * 64);
      unsigned lav = (unsigned)(size_t)(void*)xif;  // LDS byte offset (addr[31:0])
      u32x4 g0;
      g0[0] = 1u;                                          // count=1, user mode
      g0[1] = lav;                                         // lds_addr
      g0[2] = (unsigned)ga;                                // global_addr[31:0]
      g0[3] = ((unsigned)(ga >> 32) & 0x01FFFFFFu) | 0x80000000u;  // [56:32] | type=2
      i32x8 g1;
      g1[0] = (int)(2u << 16);     // data_size = 4B; no multicast / iterate / pad
      g1[1] = (int)(64u << 16);    // tensor_dim0 = 64
      g1[2] = (int)(64u << 16);    // tensor_dim1 = 64
      g1[3] = (int)(64u << 16);    // tile_dim0 = 64
      g1[4] = (int)64;             // tile_dim1 = 64, tile_dim2 = 0
      g1[5] = (int)(T_ * 64);      // tensor_dim0_stride = 4096
      g1[6] = 0;
      g1[7] = 0;
      i32x4 gz = {0, 0, 0, 0};
#if defined(__clang_major__) && __clang_major__ >= 23
      i32x8 gz8 = {0, 0, 0, 0, 0, 0, 0, 0};
      __builtin_amdgcn_tensor_load_to_lds(g0, g1, gz, gz, gz8, 0);
#else
      __builtin_amdgcn_tensor_load_to_lds(g0, g1, gz, gz, 0);
#endif
      __builtin_amdgcn_s_wait_tensorcnt(0);
    }
#else
    for (int e = tid; e < BM * 64; e += 256) {
      int r = e >> 6;
      xif[e] = data[(size_t)(row0 + r) * (T_ * 64) + (63 - i) * 64 + (e & 63)];
    }
#endif
    __syncthreads();
    for (int e = tid; e < BM * 64; e += 256) xib[e] = tobf(xif[e]);
    if (tid < BM) {
      float s = 0.f;
      for (int c = 0; c < 32; ++c) s += xif[tid * 64 + 32 + c];
      rowm[tid] = (s > 0.f) ? 1.f : 0.f;
    }

    // ---- two coupling flow layers ----
    for (int l = 0; l < 2; ++l) {
      const float* w0f = l ? f1w0 : f0w0;
      const float* b0 = l ? f1b0 : f0b0;
      const float* b1 = l ? f1b1 : f0b1;
      const float* b2 = l ? f1b2 : f0b2;
      const float* tw = l ? f1tw : f0tw;
      const bf16* W0 = wsb + (l ? OFF_F1W0 : OFF_F0W0);
      const bf16* W1 = wsb + (l ? OFF_F1W1 : OFF_F0W1);
      const bf16* W2 = wsb + (l ? OFF_F1W2 : OFF_F0W2);
      if (tid == 0) {
        __builtin_prefetch(W1, 0, 3);
        __builtin_prefetch(W2, 0, 3);
      }
      // per-step bias: fold dt column of w0; TimeTanh vector
      if (tid < HIDN) {
        bias0v[tid] = b0[tid] + dt * w0f[tid * 129 + 128];
        tvec[tid] = fast_tanh(dt * tw[tid]);
      }
      // Xm = hs * mask  (bf16 A operand)
      for (int e = tid; e < BM * RECN; e += 256) {
        int c = e & 127;
        float mc = (l == 0) ? (c < 64 ? 1.f : 0.f) : (c >= 64 ? 1.f : 0.f);
        abuf[(e >> 7) * HIDN + c] = tobf(hs[e] * mc);
      }
      __syncthreads();
      // h1 = tanh(Xm @ W0k^T + bias0(dt))
      gemm_bias_tanh(abuf, HIDN, W0, RECN, bias0v, obuf, HIDN, 16, 64, wave, lane);
      __syncthreads();
      // h2 = tanh(h1 @ W1^T + b1)
      gemm_bias_tanh(obuf, HIDN, W1, HIDN, b1, abuf, HIDN, 16, 64, wave, lane);
      __syncthreads();
      // o = h2 @ W2^T + b2 ; coupling update of hs (scale/shift tile pair)
      for (int t = wave; t < 32; t += NW) {
        int mt = t >> 3, p = t & 7;
        int cs = p * 16 + nl;           // channel 0..127
        float bS = b2[cs], bH = b2[cs + 128];
        v8f aS, aH;
#pragma unroll
        for (int v = 0; v < 8; ++v) { aS[v] = bS; aH[v] = bH; }
#pragma unroll
        for (int kb = 0; kb < HIDN; kb += 32) {
          bf16x16 av = ldsA(abuf + (mt * 16 + nl) * HIDN + kb + hi * 16);
          bf16x16 bS16 = gldB(W2 + cs * HIDN + kb + hi * 16);
          bf16x16 bH16 = gldB(W2 + (cs + 128) * HIDN + kb + hi * 16);
          aS = wmma_bf(av, bS16, aS);
          aH = wmma_bf(av, bH16, aH);
        }
        float tsv = tvec[cs] * 1.44269504088896f;   // fold log2e into exp
        float tshv = tvec[cs + 128];
        float mc = (l == 0) ? (cs < 64 ? 1.f : 0.f) : (cs >= 64 ? 1.f : 0.f);
#pragma unroll
        for (int v = 0; v < 8; ++v) {
          int r = mt * 16 + v + hi * 8;
          float x = hs[r * RECN + cs];
          float upd = x * fexp2(aS[v] * tsv) + aH[v] * tshv;
          hs[r * RECN + cs] = mc * x + (1.f - mc) * upd;
        }
      }
      __syncthreads();
    }

    // ---- GRU cell (gates tile-local in accumulators) ----
    for (int e = tid; e < BM * RECN; e += 256)            // Hf bf16
      abuf[(e >> 7) * HIDN + (e & 127)] = tobf(hs[e]);
    __syncthreads();
    {
      const bf16* Wih = wsb + OFF_WIH;
      const bf16* Whh = wsb + OFF_WHH;
      for (int t = wave; t < 32; t += NW) {
        int mt = t >> 3, nt = t & 7;
        int c = nt * 16 + nl;
        v8f ar, az, ain, ahn;
        float br = gbih[c] + gbhh[c];
        float bz = gbih[128 + c] + gbhh[128 + c];
        float bi = gbih[256 + c], bh = gbhh[256 + c];
#pragma unroll
        for (int v = 0; v < 8; ++v) { ar[v] = br; az[v] = bz; ain[v] = bi; ahn[v] = bh; }
#pragma unroll
        for (int kb = 0; kb < 64; kb += 32) {             // x_i @ Wih^T (K=64)
          bf16x16 av = ldsA(xib + (mt * 16 + nl) * 64 + kb + hi * 16);
          ar = wmma_bf(av, gldB(Wih + c * 64 + kb + hi * 16), ar);
          az = wmma_bf(av, gldB(Wih + (128 + c) * 64 + kb + hi * 16), az);
          ain = wmma_bf(av, gldB(Wih + (256 + c) * 64 + kb + hi * 16), ain);
        }
#pragma unroll
        for (int kb = 0; kb < RECN; kb += 32) {           // hf @ Whh^T (K=128)
          bf16x16 av = ldsA(abuf + (mt * 16 + nl) * HIDN + kb + hi * 16);
          ar = wmma_bf(av, gldB(Whh + c * RECN + kb + hi * 16), ar);
          az = wmma_bf(av, gldB(Whh + (128 + c) * RECN + kb + hi * 16), az);
          ahn = wmma_bf(av, gldB(Whh + (256 + c) * RECN + kb + hi * 16), ahn);
        }
#pragma unroll
        for (int v = 0; v < 8; ++v) {
          int r = mt * 16 + v + hi * 8;
          float rg = fast_sigmoid(ar[v]), zg = fast_sigmoid(az[v]);
          float ng = fast_tanh(ain[v] + rg * ahn[v]);
          float hf = hs[r * RECN + c];
          float hp = (1.f - zg) * ng + zg * hf;
          float mr = rowm[r];
          hs[r * RECN + c] = mr * hp + (1.f - mr) * hf;
        }
      }
    }
    __syncthreads();
  }

  // ---- z0 head: hid = tanh(h @ z0w0^T) ; out = hid @ z0w1^T ----
  for (int e = tid; e < BM * RECN; e += 256)
    abuf[(e >> 7) * HIDN + (e & 127)] = tobf(hs[e]);
  for (int e = tid; e < BM * 128; e += 256) obuf[e] = (bf16)0.0f;  // zero-pad hid
  __syncthreads();
  for (int t = wave; t < 28; t += NW) {                   // N=112 (padded), K=128
    int mt = t / 7, nt = t - mt * 7;
    int col = nt * 16 + nl;
    float b = (col < 100) ? zb0[col] : 0.f;
    v8f acc;
#pragma unroll
    for (int v = 0; v < 8; ++v) acc[v] = b;
#pragma unroll
    for (int kb = 0; kb < RECN; kb += 32) {
      bf16x16 av = ldsA(abuf + (mt * 16 + nl) * HIDN + kb + hi * 16);
      acc = wmma_bf(av, gldB(wsb + OFF_ZW0 + col * 128 + kb + hi * 16), acc);
    }
#pragma unroll
    for (int v = 0; v < 8; ++v) {
      int r = mt * 16 + v + hi * 8;
      obuf[r * 128 + col] = tobf(fast_tanh(acc[v]));
    }
  }
  __syncthreads();
  for (int t = wave; t < 32; t += NW) {                   // N=128, K=100 padded to 128
    int mt = t >> 3, nt = t & 7;
    int col = nt * 16 + nl;
    float b = zb1[col];
    v8f acc;
#pragma unroll
    for (int v = 0; v < 8; ++v) acc[v] = b;
#pragma unroll
    for (int kb = 0; kb < 128; kb += 32) {
      bf16x16 av = ldsA(obuf + (mt * 16 + nl) * 128 + kb + hi * 16);
      acc = wmma_bf(av, gldB(wsb + OFF_ZW1 + col * 128 + kb + hi * 16), acc);
    }
#pragma unroll
    for (int v = 0; v < 8; ++v) {
      int rg = row0 + mt * 16 + v + hi * 8;
      float x = acc[v];
      if (col < 64) out[(size_t)rg * 64 + col] = x;                         // mean
      else out[(size_t)B_ * 64 + (size_t)rg * 64 + (col - 64)] = fast_softplus(x);
    }
  }
}

extern "C" void kernel_launch(void* const* d_in, const int* in_sizes, int n_in,
                              void* d_out, int out_size, void* d_ws, size_t ws_size,
                              hipStream_t stream) {
  const float* data = (const float*)d_in[0];
  const float* ts   = (const float*)d_in[1];
  const float* f0w0 = (const float*)d_in[2];
  const float* f0b0 = (const float*)d_in[3];
  const float* f0w1 = (const float*)d_in[4];
  const float* f0b1 = (const float*)d_in[5];
  const float* f0w2 = (const float*)d_in[6];
  const float* f0b2 = (const float*)d_in[7];
  const float* f0tw = (const float*)d_in[8];
  const float* f1w0 = (const float*)d_in[9];
  const float* f1b0 = (const float*)d_in[10];
  const float* f1w1 = (const float*)d_in[11];
  const float* f1b1 = (const float*)d_in[12];
  const float* f1w2 = (const float*)d_in[13];
  const float* f1b2 = (const float*)d_in[14];
  const float* f1tw = (const float*)d_in[15];
  const float* gwih = (const float*)d_in[16];
  const float* gwhh = (const float*)d_in[17];
  const float* gbih = (const float*)d_in[18];
  const float* gbhh = (const float*)d_in[19];
  const float* zw0  = (const float*)d_in[20];
  const float* zb0  = (const float*)d_in[21];
  const float* zw1  = (const float*)d_in[22];
  const float* zb1  = (const float*)d_in[23];
  (void)in_sizes; (void)n_in; (void)out_size; (void)ws_size;

  bf16* wsb = (bf16*)d_ws;
  auto cv = [&](int off, const float* src, int sr, int sc, int sstr, int dr, int dc) {
    int n = dr * dc;
    cvt_pad<<<(n + 255) / 256, 256, 0, stream>>>(wsb + off, src, sr, sc, sstr, dr, dc);
  };
  // f32 -> bf16 weight conversion / padding into workspace
  cv(OFF_F0W0, f0w0, 256, 128, 129, 256, 128);
  cv(OFF_F0W1, f0w1, 256, 256, 256, 256, 256);
  cv(OFF_F0W2, f0w2, 256, 256, 256, 256, 256);
  cv(OFF_F1W0, f1w0, 256, 128, 129, 256, 128);
  cv(OFF_F1W1, f1w1, 256, 256, 256, 256, 256);
  cv(OFF_F1W2, f1w2, 256, 256, 256, 256, 256);
  cv(OFF_WIH,  gwih, 384, 64, 64, 384, 64);
  cv(OFF_WHH,  gwhh, 384, 128, 128, 384, 128);
  cv(OFF_ZW0,  zw0, 100, 128, 128, 112, 128);
  cv(OFF_ZW1,  zw1, 128, 100, 100, 128, 128);

  nfrnn_main<<<B_ / BM, 256, 0, stream>>>(
      data, ts, f0w0, f0b0, f0b1, f0b2, f0tw, f1w0, f1b0, f1b1, f1b2, f1tw,
      gbih, gbhh, zb0, zb1, wsb, (float*)d_out);
}
